// VectorQuantizer_31001073943027
// MI455X (gfx1250) — matrix operand order repaired and checked by reading, not compiled, not run on hardware
//
#include <hip/hip_runtime.h>
#include <hip/hip_bf16.h>

// ---------------------------------------------------------------------------
// VQ-VAE vector quantizer for MI455X (gfx1250, wave32, WMMA).
//   B = 32768 z-rows, K = 8192 codes, D = 512.
// Distance GEMM (275 GFLOP) runs on v_wmma_f32_16x16x32_bf16 with fp32
// accumulation; norms / gather / EMA stay in exact fp32. Each wave keeps a
// 32x512 A-slice resident in VGPRs so every codebook (B) fragment fetched
// from the L2-resident bf16 codebook feeds two WMMAs -> half the L2 traffic
// of the naive scheme. All reductions are deterministic (no atomics).
// ---------------------------------------------------------------------------

#define NUM_CODES 8192
#define CODE_DIM 512
#define NUM_Z 32768
#define EMA_DECAY 0.99f
#define ONE_MINUS_DECAY 0.01f
#define EPS 1e-5f
#define COMMITMENT_COST 0.25f

typedef __attribute__((ext_vector_type(16))) __bf16 bf16x16;
typedef __attribute__((ext_vector_type(8)))  float  floatx8;

// --- Kernel 1: codebook fp32 -> bf16 (row-major, D contiguous) + ||c||^2 ----
__global__ __launch_bounds__(256) void vq_prep(const float* __restrict__ cb,
                                               __bf16* __restrict__ cbbf,
                                               float* __restrict__ cnorm) {
  const int k = blockIdx.x;          // one code per block
  const int t = threadIdx.x;         // 256 threads, 2 dims each
  const float* row = cb + (size_t)k * CODE_DIM;
  float s = 0.0f;
#pragma unroll
  for (int ss = 0; ss < 2; ++ss) {
    const int d = t + ss * 256;
    const float v = row[d];
    s += v * v;
    cbbf[(size_t)k * CODE_DIM + d] = (__bf16)v;
  }
  __shared__ float red[256];
  red[t] = s;
  __syncthreads();
  for (int off = 128; off > 0; off >>= 1) {
    if (t < off) red[t] += red[t + off];
    __syncthreads();
  }
  if (t == 0) cnorm[k] = red[0];
}

// --- Kernel 2: fused distance-GEMM + per-row argmin (the WMMA kernel) ------
// Grid: 128 blocks x 256 threads (8 waves). Each wave owns 32 z-rows as two
// resident 16x512 A slices (2 x 16 bf16 fragments = 256 VGPRs) and streams
// the entire bf16 codebook (8 MB, L2-resident) as B fragments; each B
// fragment is consumed by two WMMAs (one per A slice).
// score(m,n) = ||c_n||^2 - 2 * dot(z_m, c_n)   (||z||^2 is row-constant)
__global__ __launch_bounds__(256) void vq_argmin_wmma(
    const float* __restrict__ z, const __bf16* __restrict__ cbbf,
    const float* __restrict__ cnorm, int* __restrict__ bestIdx) {
  const int wave    = threadIdx.x >> 5;
  const int lane    = threadIdx.x & 31;
  const int laneRow = lane & 15;   // M row (A) / N column (B) within tile
  const int laneHi  = lane >> 4;   // which K half-stripe this lane holds
  const int mbase   = blockIdx.x * 256 + wave * 32;

  // ---- Load resident A fragments (2 slices x 16 rows x K=512), fp32->bf16.
  // A 16x32 layout (matches B): lanes 0-15 hold K 0..15,
  // lanes 16-31 hold K 16..31  => one contiguous 16-float chunk.
  bf16x16 afrag[2][16];
#pragma unroll
  for (int s = 0; s < 2; ++s) {
    const float* zp = z + (size_t)(mbase + s * 16 + laneRow) * CODE_DIM;
#pragma unroll
    for (int kf = 0; kf < 16; ++kf) {
      const int k0 = kf * 32 + laneHi * 16;
      const float4 f0 = *(const float4*)(zp + k0);
      const float4 f1 = *(const float4*)(zp + k0 + 4);
      const float4 f2 = *(const float4*)(zp + k0 + 8);
      const float4 f3 = *(const float4*)(zp + k0 + 12);
      bf16x16 a;
      a[0]  = (__bf16)f0.x; a[1]  = (__bf16)f0.y; a[2]  = (__bf16)f0.z; a[3]  = (__bf16)f0.w;
      a[4]  = (__bf16)f1.x; a[5]  = (__bf16)f1.y; a[6]  = (__bf16)f1.z; a[7]  = (__bf16)f1.w;
      a[8]  = (__bf16)f2.x; a[9]  = (__bf16)f2.y; a[10] = (__bf16)f2.z; a[11] = (__bf16)f2.w;
      a[12] = (__bf16)f3.x; a[13] = (__bf16)f3.y; a[14] = (__bf16)f3.z; a[15] = (__bf16)f3.w;
      afrag[s][kf] = a;
    }
  }

  float best[2][8];
  int   bidx[2][8];
#pragma unroll
  for (int s = 0; s < 2; ++s)
#pragma unroll
    for (int j = 0; j < 8; ++j) { best[s][j] = 3.4e38f; bidx[s][j] = 0; }

  // B 32x16 layout used here (same K order as A): lane = column N (code), lanes 0-15 hold
  // K 0..15, lanes 16-31 hold K 16..31 => one contiguous 32-byte chunk
  // of codebook row (nb + laneRow).
  const __bf16* bbase = cbbf + (size_t)laneRow * CODE_DIM + laneHi * 16;

  for (int nb = 0; nb < NUM_CODES; nb += 16) {
    floatx8 c0 = {0.f, 0.f, 0.f, 0.f, 0.f, 0.f, 0.f, 0.f};
    floatx8 c1 = {0.f, 0.f, 0.f, 0.f, 0.f, 0.f, 0.f, 0.f};
    const float cn = cnorm[nb + laneRow];
    const __bf16* bp = bbase + (size_t)nb * CODE_DIM;
#pragma unroll
    for (int kf = 0; kf < 16; ++kf) {
      const bf16x16 b = *(const bf16x16*)(bp + kf * 32);
      c0 = __builtin_amdgcn_wmma_f32_16x16x32_bf16(
          false, afrag[0][kf], false, b, (short)0, c0, false, false);
      c1 = __builtin_amdgcn_wmma_f32_16x16x32_bf16(
          false, afrag[1][kf], false, b, (short)0, c1, false, false);
    }
    const int n = nb + laneRow;  // C layout: N = lane%16 for every acc VGPR
#pragma unroll
    for (int j = 0; j < 8; ++j) {
      const float sc0 = __builtin_fmaf(-2.0f, c0[j], cn);
      if (sc0 < best[0][j]) { best[0][j] = sc0; bidx[0][j] = n; }
      const float sc1 = __builtin_fmaf(-2.0f, c1[j], cn);
      if (sc1 < best[1][j]) { best[1][j] = sc1; bidx[1][j] = n; }
    }
  }

  // Reduce argmin across the 16 lanes that share a C row; tie-break to the
  // lowest index to match jnp.argmin. C layout: VGPR j -> M = j + 8*laneHi.
#pragma unroll
  for (int s = 0; s < 2; ++s) {
#pragma unroll
    for (int j = 0; j < 8; ++j) {
      float b = best[s][j];
      int   i = bidx[s][j];
#pragma unroll
      for (int m = 1; m < 16; m <<= 1) {
        const float ob = __shfl_xor(b, m, 32);
        const int   oi = __shfl_xor(i, m, 32);
        if (ob < b || (ob == b && oi < i)) { b = ob; i = oi; }
      }
      if (laneRow == 0) bestIdx[mbase + s * 16 + j + 8 * laneHi] = i;
    }
  }
}

// --- Kernel 3: gather z_q, per-row squared-error partial, indices ----------
__global__ __launch_bounds__(256) void vq_gather(
    const float* __restrict__ z, const float* __restrict__ cb,
    const int* __restrict__ idx, float* __restrict__ out_zq,
    float* __restrict__ out_idx, float* __restrict__ lossP) {
  const int row = blockIdx.x;
  const int t   = threadIdx.x;
  const int k   = idx[row];
  const float* zr = z + (size_t)row * CODE_DIM;
  const float* cr = cb + (size_t)k * CODE_DIM;
  float a = 0.0f;
#pragma unroll
  for (int ss = 0; ss < 2; ++ss) {
    const int d = t + ss * 256;
    const float zv = zr[d];
    const float q  = cr[d];
    out_zq[(size_t)row * CODE_DIM + d] = q;  // z + sg(z_q - z) == z_q
    const float df = zv - q;
    a += df * df;
  }
  __shared__ float red[256];
  red[t] = a;
  __syncthreads();
  for (int off = 128; off > 0; off >>= 1) {
    if (t < off) red[t] += red[t + off];
    __syncthreads();
  }
  if (t == 0) { lossP[row] = red[0]; out_idx[row] = (float)k; }
}

// --- Kernel 4: deterministic segment sums + EMA update (per code) ----------
__global__ __launch_bounds__(256) void vq_segment_ema(
    const float* __restrict__ z, const int* __restrict__ idx,
    const float* __restrict__ ema_sum, const float* __restrict__ ema_count,
    float* __restrict__ out_nsum, float* __restrict__ out_ncnt) {
  const int k = blockIdx.x;
  const int t = threadIdx.x;
  const int d0 = t, d1 = t + 256;
  __shared__ int sIdx[256];
  float acc0 = 0.0f, acc1 = 0.0f;
  int cnt = 0;
  for (int base = 0; base < NUM_Z; base += 256) {
    sIdx[t] = idx[base + t];
    __syncthreads();
    for (int j = 0; j < 256; ++j) {  // uniform branch: scans in fixed order
      if (sIdx[j] == k) {
        const float* zr = z + (size_t)(base + j) * CODE_DIM;
        acc0 += zr[d0];
        acc1 += zr[d1];
        ++cnt;
      }
    }
    __syncthreads();
  }
  out_nsum[(size_t)k * CODE_DIM + d0] =
      EMA_DECAY * ema_sum[(size_t)k * CODE_DIM + d0] + ONE_MINUS_DECAY * acc0;
  out_nsum[(size_t)k * CODE_DIM + d1] =
      EMA_DECAY * ema_sum[(size_t)k * CODE_DIM + d1] + ONE_MINUS_DECAY * acc1;
  if (t == 0)
    out_ncnt[k] = EMA_DECAY * ema_count[k] + ONE_MINUS_DECAY * (float)cnt;
}

// --- Kernel 5: single-block deterministic reductions (loss, n) -------------
__global__ __launch_bounds__(256) void vq_reduce_scalars(
    const float* __restrict__ lossP, const float* __restrict__ new_count,
    float* __restrict__ nOut, float* __restrict__ out_loss) {
  const int t = threadIdx.x;
  float ls = 0.0f;
  for (int i = t; i < NUM_Z; i += 256) ls += lossP[i];
  float ns = 0.0f;
  for (int i = t; i < NUM_CODES; i += 256) ns += new_count[i];
  __shared__ float r1[256], r2[256];
  r1[t] = ls; r2[t] = ns;
  __syncthreads();
  for (int off = 128; off > 0; off >>= 1) {
    if (t < off) { r1[t] += r1[t + off]; r2[t] += r2[t + off]; }
    __syncthreads();
  }
  if (t == 0) {
    *nOut = r2[0];
    *out_loss = COMMITMENT_COST * r1[0] / ((float)NUM_Z * (float)CODE_DIM);
  }
}

// --- Kernel 6: new_codebook = new_sum / count_smoothed ---------------------
__global__ __launch_bounds__(256) void vq_finalize_cb(
    const float* __restrict__ new_sum, const float* __restrict__ new_count,
    const float* __restrict__ nScal, float* __restrict__ out_cb) {
  const int k = blockIdx.x;
  const int t = threadIdx.x;
  const float n = *nScal;
  const float denom = (new_count[k] + EPS) / (n + (float)NUM_CODES * EPS) * n;
  const float inv = 1.0f / denom;
#pragma unroll
  for (int ss = 0; ss < 2; ++ss) {
    const int d = t + ss * 256;
    out_cb[(size_t)k * CODE_DIM + d] = new_sum[(size_t)k * CODE_DIM + d] * inv;
  }
}

// ---------------------------------------------------------------------------
extern "C" void kernel_launch(void* const* d_in, const int* in_sizes, int n_in,
                              void* d_out, int out_size, void* d_ws,
                              size_t ws_size, hipStream_t stream) {
  (void)in_sizes; (void)n_in; (void)out_size; (void)ws_size;
  const float* z         = (const float*)d_in[0];
  const float* cb        = (const float*)d_in[1];
  const float* ema_sum   = (const float*)d_in[2];
  const float* ema_count = (const float*)d_in[3];
  float* out = (float*)d_out;

  // Workspace (~8.7 MB): bf16 codebook, code norms, argmin indices,
  // per-row loss partials, scalar n.
  char* ws = (char*)d_ws;
  __bf16* cbbf  = (__bf16*)ws;                                   // 8 MB
  float*  cnorm = (float*)(ws + 8u * 1024 * 1024);               // 32 KB
  int*    bidx  = (int*)(ws + 8u * 1024 * 1024 + 32 * 1024);     // 128 KB
  float*  lossP = (float*)(ws + 8u * 1024 * 1024 + 160 * 1024);  // 128 KB
  float*  nScal = (float*)(ws + 8u * 1024 * 1024 + 288 * 1024);  // 4 B

  // Flat output layout (reference return order).
  float* out_zq   = out;             // 32768*512
  float* out_loss = out + 16777216;  // 1
  float* out_idx  = out + 16777217;  // 32768 (indices as float)
  float* out_ncb  = out + 16809985;  // 8192*512
  float* out_ncnt = out + 21004289;  // 8192
  float* out_nsum = out + 21012481;  // 8192*512

  vq_prep<<<NUM_CODES, 256, 0, stream>>>(cb, cbbf, cnorm);
  vq_argmin_wmma<<<NUM_Z / 256, 256, 0, stream>>>(z, cbbf, cnorm, bidx);
  vq_gather<<<NUM_Z, 256, 0, stream>>>(z, cb, bidx, out_zq, out_idx, lossP);
  vq_segment_ema<<<NUM_CODES, 256, 0, stream>>>(z, bidx, ema_sum, ema_count,
                                                out_nsum, out_ncnt);
  vq_reduce_scalars<<<1, 256, 0, stream>>>(lossP, out_ncnt, nScal, out_loss);
  vq_finalize_cb<<<NUM_CODES, 256, 0, stream>>>(out_nsum, out_ncnt, nScal,
                                                out_ncb);
}
